// DFMP_6717328851803
// MI455X (gfx1250) — matrix operand order repaired and checked by reading, not compiled, not run on hardware
//
#include <hip/hip_runtime.h>
#include <hip/hip_bf16.h>
#include <math.h>

// ---------------------------------------------------------------------------
// Problem constants
// ---------------------------------------------------------------------------
#define B_    256
#define LR_   1024
#define G_    512
#define K_    64
#define D_    256
#define D2_   512
#define T_    8

typedef __attribute__((ext_vector_type(16))) _Float16 v16h;
typedef __attribute__((ext_vector_type(8)))  _Float16 v8h;
typedef __attribute__((ext_vector_type(8)))  float    v8f;

// ---------------------------------------------------------------------------
// Shared-memory layout for the main kernel (bytes)
// ---------------------------------------------------------------------------
#define OFF_FTA   0u         /* 64x256 f32  = 65536 */
#define OFF_FTB   65536u     /* 64x256 f32  = 65536 */
#define OFF_FTH   131072u    /* 64x256 f16  = 32768 */
#define OFF_XH    163840u    /* 64x256 f16  = 32768 */
#define OFF_H1    196608u    /* 64x512 f16  = 65536 */
#define OFF_UI    262144u    /* 256 f32 */
#define OFF_UG    263168u
#define OFF_SI    264192u    /* 64 f32 */
#define OFF_SG    264448u
#define OFF_EI    264704u
#define OFF_EG    264960u
#define OFF_PI    265216u
#define OFF_PG    265472u
#define OFF_HI    265728u    /* 256 f32 */
#define OFF_HG    266752u
#define OFF_RTI   267776u    /* 512 f32 */
#define OFF_RTG   269824u
#define OFF_RR    271872u    /* 256 f32 */
#define OFF_GF    272896u
#define OFF_GI    273920u
#define SMEM_BYTES 274944u

// ---------------------------------------------------------------------------
// Fragment helpers (f16 16x16x32 WMMA)
// ---------------------------------------------------------------------------
// A fragment: 16x32 f16, row-major LDS source with given row stride (halves).
// Lane 0..15 : row M = lane,       K = k0 + {0..7, 16..23}
// Lane 16..31: row M = lane-16,    K = k0 + {8..15, 24..31}
__device__ __forceinline__ v16h load_a_frag(const _Float16* base, int m0, int k0,
                                            int stride, int lane) {
  int row = m0 + (lane & 15);
  int kb  = k0 + ((lane >> 4) << 3);
  const _Float16* p = base + row * stride + kb;
  union { v16h v; struct { v8h lo, hi; } s; } u;
  u.s.lo = *(const v8h*)(p);
  u.s.hi = *(const v8h*)(p + 16);
  return u.v;
}

// B fragment from fragment-packed weights: each 16(N)x32(K) tile is a
// contiguous 1KB block in lane-major order; a wave reads 1KB fully coalesced.
// Block index = nt * nks + ks;  lane slot = 16 halves (32B).
__device__ __forceinline__ v16h load_b_packed(const _Float16* wp, int nt, int ks,
                                              int nks, int lane) {
  return *(const v16h*)(wp + (size_t)(((nt * nks + ks) << 5) + lane) * 16);
}

__device__ __forceinline__ v8f wmma_f16(v16h a, v16h b, v8f c) {
  return __builtin_amdgcn_wmma_f32_16x16x32_f16(false, a, false, b,
                                                (short)0, c, false, false);
}

// ---------------------------------------------------------------------------
// Fuzzy helpers
// ---------------------------------------------------------------------------
__device__ __forceinline__ float mu_s_unit(float x) {
  const float den = 2.0f + 1e-12f;
  if (x > 1.0f) return 1.0f;
  if (x > 0.0f) { float q = (1.0f - x) / den; return 1.0f - 2.0f * q * q; }
  if (x > -1.0f) { float q = (x + 1.0f) / den; return 2.0f * q * q; }
  return 0.0f;
}

__device__ __forceinline__ float centroid_bell(float c, float ab, float bb) {
  float a = fmaxf(ab, 0.001f);
  float b = fmaxf(bb, 0.001f);
  float num = 0.0f, den = 0.0f;
  #pragma unroll 1
  for (int j = 0; j < 41; ++j) {
    float g  = -4.0f + 0.2f * (float)j;
    float z  = fabsf((g - c) / a);
    float mu = 1.0f / (1.0f + powf(z, 2.0f * b));
    num += g * mu; den += mu;
  }
  return num / (den + 1e-12f);
}

__device__ __forceinline__ float centroid_s(float c) {
  float a2 = fminf(fmaxf(c - 1.0f, -4.0f), 4.0f);
  float b2 = fminf(fmaxf(c + 1.0f, -4.0f), 4.0f);
  float mid = 0.5f * (a2 + b2);
  float den = b2 - a2 + 1e-12f;
  float num = 0.0f, ds = 0.0f;
  #pragma unroll 1
  for (int j = 0; j < 41; ++j) {
    float g = -4.0f + 0.2f * (float)j;
    float y;
    if (g > b2)       { y = 1.0f; }
    else if (g > mid) { float q = (b2 - g) / den; y = 1.0f - 2.0f * q * q; }
    else if (g > a2)  { float q = (g - a2) / den; y = 2.0f * q * q; }
    else              { y = 0.0f; }
    num += g * y; ds += y;
  }
  return num / (ds + 1e-12f);
}

__device__ __forceinline__ float defuzz_pair(float rb_, float rs_, float ab, float bb) {
  float cb = 4.0f * tanhf(rb_);
  float cs = 4.0f * tanhf(rs_);
  return 0.5f * centroid_bell(cb, ab, bb) + 0.5f * centroid_s(cs);
}

// ---------------------------------------------------------------------------
// Prep kernel 1: weight f32 row-major (Krows x Ncols) -> f16 fragment-packed.
// Packed element gid maps to: j = K offset within lane slot, lane = frag lane,
// blk = nt * (Krows/32) + ks.  Matches load_b_packed exactly.
// ---------------------------------------------------------------------------
__global__ __launch_bounds__(256)
void k_wpack(const float* __restrict__ src, _Float16* __restrict__ dst,
             int krows, int ncols) {
  int gid = blockIdx.x * blockDim.x + threadIdx.x;
  if (gid >= krows * ncols) return;
  int nks  = krows >> 5;
  int j    = gid & 15;
  int lane = (gid >> 4) & 31;
  int blk  = gid >> 9;
  int ks   = blk % nks;
  int nt   = blk / nks;
  int n = (nt << 4) + (lane & 15);
  int k = (ks << 5) + ((lane >> 4) << 3) + (j & 7) + ((j >> 3) << 4);
  dst[gid] = (_Float16)src[k * ncols + n];
}

// ---------------------------------------------------------------------------
// Prep kernel 2: lr/g projection + fuzzify -> lr_f (B x 512), g_f (B x 512)
// ---------------------------------------------------------------------------
__global__ __launch_bounds__(256)
void k_embed(const float* __restrict__ wsi, const float* __restrict__ gen,
             const float* __restrict__ W_lr, const float* __restrict__ b_lr,
             const float* __restrict__ W_g,  const float* __restrict__ b_g,
             const float* __restrict__ abp,  const float* __restrict__ bbp,
             float* __restrict__ lr_f, float* __restrict__ g_f) {
  int gid = blockIdx.x * blockDim.x + threadIdx.x;
  if (gid >= B_ * D_) return;
  int b = gid >> 8, d = gid & 255;
  float ab = abp[0], bb = bbp[0];
  float aa = fmaxf(ab, 0.001f), bbv = fmaxf(bb, 0.001f);

  float acc = b_lr[d];
  for (int e = 0; e < LR_; ++e) acc += wsi[b * LR_ + e] * W_lr[e * D_ + d];
  float xs = 4.0f * tanhf(acc * 0.25f);
  lr_f[b * D2_ + d]       = 1.0f / (1.0f + powf(fabsf(xs / aa), 2.0f * bbv));
  lr_f[b * D2_ + D_ + d]  = mu_s_unit(xs);

  float acc2 = b_g[d];
  for (int e = 0; e < G_; ++e) acc2 += gen[b * G_ + e] * W_g[e * D_ + d];
  float xs2 = 4.0f * tanhf(acc2 * 0.25f);
  g_f[b * D2_ + d]        = 1.0f / (1.0f + powf(fabsf(xs2 / aa), 2.0f * bbv));
  g_f[b * D2_ + D_ + d]   = mu_s_unit(xs2);
}

// ---------------------------------------------------------------------------
// Prep kernel 3: QI = lr_f @ WqI, QG = g_f @ WqG
// ---------------------------------------------------------------------------
__global__ __launch_bounds__(256)
void k_q(const float* __restrict__ lr_f, const float* __restrict__ g_f,
         const float* __restrict__ WqI,  const float* __restrict__ WqG,
         float* __restrict__ QI, float* __restrict__ QG) {
  int gid = blockIdx.x * blockDim.x + threadIdx.x;
  if (gid >= B_ * D_) return;
  int b = gid >> 8, d = gid & 255;
  float aI = 0.0f, aG = 0.0f;
  for (int e = 0; e < D2_; ++e) {
    aI += lr_f[b * D2_ + e] * WqI[e * D_ + d];
    aG += g_f [b * D2_ + e] * WqG[e * D_ + d];
  }
  QI[gid] = aI; QG[gid] = aG;
}

// ---------------------------------------------------------------------------
// Prep kernel 4: uI = WkI @ QI[b]  (factored attention key vector)
// ---------------------------------------------------------------------------
__global__ __launch_bounds__(256)
void k_u(const float* __restrict__ QI, const float* __restrict__ QG,
         const float* __restrict__ WkI, const float* __restrict__ WkG,
         float* __restrict__ UI, float* __restrict__ UG) {
  int gid = blockIdx.x * blockDim.x + threadIdx.x;
  if (gid >= B_ * D_) return;
  int b = gid >> 8, e = gid & 255;
  const float* qi = QI + b * D_;
  const float* qg = QG + b * D_;
  const float* wi = WkI + e * D_;
  const float* wg = WkG + e * D_;
  float aI = 0.0f, aG = 0.0f;
  for (int d = 0; d < D_; ++d) { aI += wi[d] * qi[d]; aG += wg[d] * qg[d]; }
  UI[gid] = aI; UG[gid] = aG;
}

// ---------------------------------------------------------------------------
// Main recurrent kernel: one block per batch element; Ft resident in LDS.
// ---------------------------------------------------------------------------
__global__ __launch_bounds__(256)
void k_main(const float* __restrict__ FT,
            const float* __restrict__ WvI, const float* __restrict__ WvG,
            const float* __restrict__ b1,  const float* __restrict__ b2,
            const float* __restrict__ Wf,  const float* __restrict__ bfv,
            const float* __restrict__ Wi,  const float* __restrict__ biv,
            const float* __restrict__ abp, const float* __restrict__ bbp,
            const float* __restrict__ UI,  const float* __restrict__ UG,
            const _Float16* __restrict__ W1p, const _Float16* __restrict__ W2p,
            const _Float16* __restrict__ Vfp, const _Float16* __restrict__ Vip,
            float* __restrict__ out) {
  extern __shared__ __align__(16) char smem[];
  float*    FtA = (float*)(smem + OFF_FTA);
  float*    FtB = (float*)(smem + OFF_FTB);
  _Float16* Fth = (_Float16*)(smem + OFF_FTH);
  _Float16* Xh  = (_Float16*)(smem + OFF_XH);
  _Float16* H1h = (_Float16*)(smem + OFF_H1);
  float* uI  = (float*)(smem + OFF_UI);
  float* uG  = (float*)(smem + OFF_UG);
  float* sI  = (float*)(smem + OFF_SI);
  float* sG  = (float*)(smem + OFF_SG);
  float* eI  = (float*)(smem + OFF_EI);
  float* eG  = (float*)(smem + OFF_EG);
  float* pI  = (float*)(smem + OFF_PI);
  float* pG  = (float*)(smem + OFF_PG);
  float* hI  = (float*)(smem + OFF_HI);
  float* hG  = (float*)(smem + OFF_HG);
  float* rtI = (float*)(smem + OFF_RTI);
  float* rtG = (float*)(smem + OFF_RTG);
  float* rr  = (float*)(smem + OFF_RR);
  float* gfb = (float*)(smem + OFF_GF);
  float* gib = (float*)(smem + OFF_GI);

  const int b    = blockIdx.x;
  const int tid  = threadIdx.x;
  const int lane = tid & 31;
  const int wv   = tid >> 5;
  const float ab = abp[0], bb = bbp[0];
  const float rscale = 0.0625f;  // 1/sqrt(D)

  // Per-batch key vectors + initial state
  uI[tid] = UI[b * D_ + tid];
  uG[tid] = UG[b * D_ + tid];
  float* Ftcur = FtA;
  float* Ftnxt = FtB;
  for (int i = tid; i < K_ * D_; i += 256) Ftcur[i] = FT[i];
  __syncthreads();

  for (int t = 0; t < T_; ++t) {
    // --- 1. attention scores: s_k = Ft[k] . u ---
    if (tid < K_) {
      const float* row = Ftcur + tid * D_;
      float aI = 0.0f, aG = 0.0f;
      for (int d = 0; d < D_; ++d) { float f = row[d]; aI += f * uI[d]; aG += f * uG[d]; }
      sI[tid] = aI * rscale;
      sG[tid] = aG * rscale;
    }
    __syncthreads();
    // --- 2. softmax over K=64 ---
    if (tid < 128) {
      const float* s = (tid < 64) ? sI : sG;
      float* e = (tid < 64) ? eI : eG;
      int k = tid & 63;
      float m = -1e30f;
      for (int j = 0; j < K_; ++j) m = fmaxf(m, s[j]);
      e[k] = expf(s[k] - m);
    }
    __syncthreads();
    if (tid < 128) {
      const float* e = (tid < 64) ? eI : eG;
      float* p = (tid < 64) ? pI : pG;
      int k = tid & 63;
      float sum = 0.0f;
      for (int j = 0; j < K_; ++j) sum += e[j];
      p[k] = e[k] / sum;
    }
    __syncthreads();
    // --- 3. h = p^T Ft  (D-vector) ---
    {
      int d = tid;
      float aI = 0.0f, aG = 0.0f;
      for (int k = 0; k < K_; ++k) {
        float f = Ftcur[k * D_ + d];
        aI += pI[k] * f; aG += pG[k] * f;
      }
      hI[d] = aI; hG[d] = aG;
    }
    __syncthreads();
    // --- 4. rt = h @ Wv  (2D-vector) ---
    for (int e2 = tid; e2 < D2_; e2 += 256) {
      float aI = 0.0f, aG = 0.0f;
      for (int d = 0; d < D_; ++d) {
        aI += hI[d] * WvI[d * D2_ + e2];
        aG += hG[d] * WvG[d * D2_ + e2];
      }
      rtI[e2] = aI; rtG[e2] = aG;
    }
    __syncthreads();
    // --- 5. defuzzify -> r (D-vector) ---
    {
      int d = tid;
      float dI = defuzz_pair(rtI[d], rtI[D_ + d], ab, bb);
      float dG = defuzz_pair(rtG[d], rtG[D_ + d], ab, bb);
      rr[d] = 0.5f * dI + 0.5f * dG;
    }
    __syncthreads();
    // --- 6. gate bias vectors: gf = r@Wf + bf, gi = r@Wi + bi ---
    {
      int d = tid;
      float af = bfv[d], ai = biv[d];
      for (int e = 0; e < D_; ++e) {
        float rv = rr[e];
        af += rv * Wf[e * D_ + d];
        ai += rv * Wi[e * D_ + d];
      }
      gfb[d] = af; gib[d] = ai;
    }
    // --- 7. build f16 mirrors: Fth = f16(Ft), Xh = f16(Ft + r) ---
    for (int i = tid; i < K_ * D_; i += 256) {
      float v = Ftcur[i];
      Fth[i] = (_Float16)v;
      Xh[i]  = (_Float16)(v + rr[i & 255]);
    }
    __syncthreads();
    // --- 8. H1 = gelu(x @ W1 + b1)  (64x512, f16 into LDS) ---
    // Wave-uniform row block: hoist the 8 A-fragments, loop over 16 N-tiles.
    {
      int m0     = (wv >> 1) * 16;
      int ntbase = (wv & 1) * 16;
      v16h a8[8];
      #pragma unroll
      for (int ks = 0; ks < 8; ++ks)
        a8[ks] = load_a_frag(Xh, m0, ks * 32, D_, lane);
      for (int i = 0; i < 16; ++i) {
        int nt = ntbase + i;
        if (i + 1 < 16)  // stream next packed weight tile toward L0/L2
          __builtin_prefetch(W1p + (size_t)((nt + 1) * 8) * 512, 0, 1);
        v8f acc = {};
        #pragma unroll
        for (int ks = 0; ks < 8; ++ks)
          acc = wmma_f16(a8[ks], load_b_packed(W1p, nt, ks, 8, lane), acc);
        int n0  = nt * 16;
        int col = n0 + (lane & 15);
        int rb0 = (lane >> 4) << 3;
        float bias = b1[col];
        #pragma unroll
        for (int r = 0; r < 8; ++r) {
          float v = acc[r] + bias;
          float g = 0.5f * v * (1.0f + erff(v * 0.70710678118f));
          H1h[(m0 + rb0 + r) * D2_ + col] = (_Float16)g;
        }
      }
    }
    __syncthreads();
    // --- 9. final tiles: F_tilde, gates, state update, output write ---
    {
      float* dst = out + ((size_t)(T_ - 1 - t) * B_ + b) * (size_t)(K_ * D_);
      int m0     = (wv >> 1) * 16;
      int ntbase = (wv & 1) * 8;
      v16h hA[16];
      #pragma unroll
      for (int ks = 0; ks < 16; ++ks)
        hA[ks] = load_a_frag(H1h, m0, ks * 32, D2_, lane);
      v16h fA[8];
      #pragma unroll
      for (int ks = 0; ks < 8; ++ks)
        fA[ks] = load_a_frag(Fth, m0, ks * 32, D_, lane);
      for (int i = 0; i < 8; ++i) {
        int nt = ntbase + i;
        if (i + 1 < 8)
          __builtin_prefetch(W2p + (size_t)((nt + 1) * 16) * 512, 0, 1);
        v8f acc2 = {}, accF = {}, accI = {};
        #pragma unroll
        for (int ks = 0; ks < 16; ++ks)
          acc2 = wmma_f16(hA[ks], load_b_packed(W2p, nt, ks, 16, lane), acc2);
        #pragma unroll
        for (int ks = 0; ks < 8; ++ks) {
          accF = wmma_f16(fA[ks], load_b_packed(Vfp, nt, ks, 8, lane), accF);
          accI = wmma_f16(fA[ks], load_b_packed(Vip, nt, ks, 8, lane), accI);
        }
        int n0   = nt * 16;
        int col  = n0 + (lane & 15);
        int rb0  = (lane >> 4) << 3;
        float b2v = b2[col];
        float rv  = rr[col];
        float gfv = gfb[col];
        float giv = gib[col];
        #pragma unroll
        for (int r = 0; r < 8; ++r) {
          int row = m0 + rb0 + r;
          float ftv  = Ftcur[row * D_ + col];
          float xv   = ftv + rv;
          float ftil = acc2[r] + b2v + xv;
          float fg = 1.0f / (1.0f + expf(-(gfv + tanhf(accF[r]))));
          float ig = 1.0f / (1.0f + expf(-(giv + tanhf(accI[r]))));
          float nv = fg * ftv + ig * tanhf(ftil);
          Ftnxt[row * D_ + col] = nv;
          dst[row * D_ + col]   = nv;
        }
      }
    }
    __syncthreads();
    float* tmp = Ftcur; Ftcur = Ftnxt; Ftnxt = tmp;
  }
}

// ---------------------------------------------------------------------------
// Host launcher
// ---------------------------------------------------------------------------
extern "C" void kernel_launch(void* const* d_in, const int* in_sizes, int n_in,
                              void* d_out, int out_size, void* d_ws, size_t ws_size,
                              hipStream_t stream) {
  const float* wsi    = (const float*)d_in[0];
  const float* gen    = (const float*)d_in[1];
  const float* FT     = (const float*)d_in[2];
  const float* W_lr   = (const float*)d_in[3];
  const float* b_lr   = (const float*)d_in[4];
  const float* W_g    = (const float*)d_in[5];
  const float* b_g    = (const float*)d_in[6];
  const float* WqI    = (const float*)d_in[7];
  const float* WqG    = (const float*)d_in[8];
  const float* WkI    = (const float*)d_in[9];
  const float* WkG    = (const float*)d_in[10];
  const float* WvI    = (const float*)d_in[11];
  const float* WvG    = (const float*)d_in[12];
  const float* mlp_W1 = (const float*)d_in[13];
  const float* mlp_b1 = (const float*)d_in[14];
  const float* mlp_W2 = (const float*)d_in[15];
  const float* mlp_b2 = (const float*)d_in[16];
  const float* Wf     = (const float*)d_in[17];
  const float* bfv    = (const float*)d_in[18];
  const float* Wi     = (const float*)d_in[19];
  const float* biv    = (const float*)d_in[20];
  const float* Vf     = (const float*)d_in[21];
  const float* Vi     = (const float*)d_in[22];
  const float* a_bell = (const float*)d_in[23];
  const float* b_bell = (const float*)d_in[24];
  float* out = (float*)d_out;

  // Workspace partition
  char* ws = (char*)d_ws;
  size_t o = 0;
  _Float16* W1p = (_Float16*)(ws + o); o += (size_t)D2_ * D_  * 2;  // 262144
  _Float16* W2p = (_Float16*)(ws + o); o += (size_t)D_  * D2_ * 2;  // 262144
  _Float16* Vfp = (_Float16*)(ws + o); o += (size_t)D_  * D_  * 2;  // 131072
  _Float16* Vip = (_Float16*)(ws + o); o += (size_t)D_  * D_  * 2;  // 131072
  float* lr_f = (float*)(ws + o); o += (size_t)B_ * D2_ * 4;        // 524288
  float* g_f  = (float*)(ws + o); o += (size_t)B_ * D2_ * 4;        // 524288
  float* QIw  = (float*)(ws + o); o += (size_t)B_ * D_ * 4;
  float* QGw  = (float*)(ws + o); o += (size_t)B_ * D_ * 4;
  float* UIw  = (float*)(ws + o); o += (size_t)B_ * D_ * 4;
  float* UGw  = (float*)(ws + o); o += (size_t)B_ * D_ * 4;
  (void)ws_size; (void)in_sizes; (void)n_in; (void)out_size;

  // Weight conversion: f32 row-major -> f16 fragment-packed (coalesced B frags)
  k_wpack<<<(D_ * D2_ + 255) / 256, 256, 0, stream>>>(mlp_W1, W1p, D_, D2_);
  k_wpack<<<(D2_ * D_ + 255) / 256, 256, 0, stream>>>(mlp_W2, W2p, D2_, D_);
  k_wpack<<<(D_ * D_  + 255) / 256, 256, 0, stream>>>(Vf, Vfp, D_, D_);
  k_wpack<<<(D_ * D_  + 255) / 256, 256, 0, stream>>>(Vi, Vip, D_, D_);

  // Front-end: embeddings, fuzzify, Q, factored key vectors u = Wk @ q
  k_embed<<<(B_ * D_ + 255) / 256, 256, 0, stream>>>(
      wsi, gen, W_lr, b_lr, W_g, b_g, a_bell, b_bell, lr_f, g_f);
  k_q<<<(B_ * D_ + 255) / 256, 256, 0, stream>>>(lr_f, g_f, WqI, WqG, QIw, QGw);
  k_u<<<(B_ * D_ + 255) / 256, 256, 0, stream>>>(QIw, QGw, WkI, WkG, UIw, UGw);

  // Main recurrent kernel: one block per batch, Ft resident in LDS
  hipFuncSetAttribute((const void*)k_main,
                      hipFuncAttributeMaxDynamicSharedMemorySize,
                      (int)SMEM_BYTES);
  k_main<<<B_, 256, SMEM_BYTES, stream>>>(
      FT, WvI, WvG, mlp_b1, mlp_b2, Wf, bfv, Wi, biv, a_bell, b_bell,
      UIw, UGw, W1p, W2p, Vfp, Vip, out);
}
